// InterpretableMultiHeadAttention_30846455120695
// MI455X (gfx1250) — compile-verified, matrix operand order checked by reading, not run on hardware
//
#include <hip/hip_runtime.h>
#include <hip/hip_bf16.h>

typedef __attribute__((ext_vector_type(16))) __bf16 v16bf;
typedef __attribute__((ext_vector_type(8)))  float  v8f;
typedef __attribute__((ext_vector_type(2)))  float  v2f;
typedef __attribute__((ext_vector_type(4)))  int    v4i;

#define EMBED    256
#define HEADS    4
#define HEAD_DIM 64
#define BATCH    4
#define SQ       512
#define SK       512
#define MROWS    (BATCH * SQ)        // 2048 rows for the big GEMMs

// Hardware tanh (CDNA5 TRANS op).
#if defined(__has_builtin) && __has_builtin(__builtin_amdgcn_tanhf)
__device__ __forceinline__ float htanh(float x) { return __builtin_amdgcn_tanhf(x); }
#else
__device__ __forceinline__ float htanh(float x) {
  float r;
  asm volatile("v_tanh_f32 %0, %1" : "=v"(r) : "v"(x));
  return r;
}
#endif

// gfx1250 async global->LDS path (ASYNCcnt-tracked), with sync fallback.
#if defined(__has_builtin) && __has_builtin(__builtin_amdgcn_global_load_async_to_lds_b128)
#define HAVE_ASYNC_LDS 1
#else
#define HAVE_ASYNC_LDS 0
#endif

#if defined(__has_builtin) && __has_builtin(__builtin_amdgcn_s_wait_asynccnt)
#define WAIT_ASYNC() __builtin_amdgcn_s_wait_asynccnt(0)
#else
#define WAIT_ASYNC() asm volatile("s_wait_asynccnt 0x0" ::: "memory")
#endif

#if HAVE_ASYNC_LDS
typedef __attribute__((address_space(1))) v4i* gas_v4i_ptr;   // global
typedef __attribute__((address_space(3))) v4i* las_v4i_ptr;   // LDS
__device__ __forceinline__ void async_b128(const float* g, float* l) {
  // Generic->AS casts via integer: global generic VA == AS1 VA; low 32 bits of
  // a generic LDS pointer are the AS3 offset on AMDGPU.
  __builtin_amdgcn_global_load_async_to_lds_b128(
      (gas_v4i_ptr)(uintptr_t)(const void*)g,
      (las_v4i_ptr)(uintptr_t)(void*)l,
      0, 0);
}
#endif

// ---------------------------------------------------------------------------
// GEMM: C[M=2048, N=256] = A[2048,256] @ W[256,256]^T + bias, via
// v_wmma_f32_16x16x32_bf16. One wave per 16x16 output tile, 4 waves/block.
// 16-bit A/B fragment layout: lane<16 holds K={0..7,16..23}+k0,
// lane>=16 holds K={8..15,24..31}+k0, as (K,K+1) pairs per VGPR.
// ---------------------------------------------------------------------------
__global__ __launch_bounds__(128) void gemm_bf16_wmma(
    const float* __restrict__ A, const float* __restrict__ W,
    const float* __restrict__ bias, float* __restrict__ C) {
  const int K = EMBED, N = EMBED;
  const int wave = threadIdx.x >> 5;
  const int lane = threadIdx.x & 31;
  const int tile = blockIdx.x * 4 + wave;       // 2048 tiles total
  const int ntiles = N / 16;                    // 16
  const int tm = (tile / ntiles) * 16;
  const int tn = (tile % ntiles) * 16;
  const int l16  = lane & 15;
  const int half = lane >> 4;

  const float* arow = A + (size_t)(tm + l16) * K;
  const float* wrow = W + (size_t)(tn + l16) * K;

  v8f c = {};
#pragma unroll
  for (int k0 = 0; k0 < K; k0 += 32) {
    v16bf fa, fb;
#pragma unroll
    for (int v = 0; v < 8; ++v) {
      const int ko = ((v < 4) ? (2 * v) : (16 + 2 * (v - 4))) + half * 8;
      const float2 pa = *(const float2*)(arow + k0 + ko);
      const float2 pb = *(const float2*)(wrow + k0 + ko);
      fa[2 * v]     = (__bf16)pa.x;
      fa[2 * v + 1] = (__bf16)pa.y;
      fb[2 * v]     = (__bf16)pb.x;
      fb[2 * v + 1] = (__bf16)pb.y;
    }
    c = __builtin_amdgcn_wmma_f32_16x16x32_bf16(false, fa, false, fb,
                                                (short)0, c, false, false);
  }
  const float bv = bias[tn + l16];
#pragma unroll
  for (int r = 0; r < 8; ++r) {
    const int m = tm + r + half * 8;            // C/D layout: VGPR r -> M=r (+8 hi half)
    C[(size_t)m * N + tn + l16] = c[r] + bv;
  }
}

// ---------------------------------------------------------------------------
// Additive scores + softmax. One wave per q row, 8 q rows per block, block
// handles one (b,h). K staged in LDS as double-buffered 64-row tiles via
// async global->LDS b128 loads (prefetch tile t+1 while computing tile t).
// Inner loop: per 4 d-values x 2 k-rows -> 4 ds_read_b128 for 8 v_tanh_f32
// + 8 FMA (0.5 DS ops per tanh; 8 independent accumulators for TRANS ILP).
// ---------------------------------------------------------------------------
__global__ __launch_bounds__(256) void scores_softmax(
    const float* __restrict__ Qb, const float* __restrict__ Kb,
    const float* __restrict__ vparam, float* __restrict__ attn) {
  __shared__ __align__(16) float  kt[2][64 * 68];   // double-buffered, bank-skewed
  __shared__ __align__(16) float2 vq[8][HEAD_DIM];  // per-wave packed (Q[d], v[d])

  const int bh   = blockIdx.x >> 6;             // 0..15
  const int qblk = blockIdx.x & 63;             // 0..63
  const int b = bh >> 2, h = bh & 3;
  const int tid  = threadIdx.x;
  const int wave = tid >> 5, lane = tid & 31;
  const int q = qblk * 8 + wave;

  const float* qrow = Qb + ((size_t)(b * SQ + q)) * EMBED + h * HEAD_DIM;
  const float* vrow = vparam + h * HEAD_DIM;
  for (int i = lane; i < HEAD_DIM; i += 32)
    vq[wave][i] = make_float2(qrow[i], vrow[i]);

  // Per-thread share of a tile: 4 x b128 (slots s = tid + i*256 of 1024).
  const float* kbase_ptr = Kb + (size_t)(b * SK) * EMBED + h * HEAD_DIM;

#if HAVE_ASYNC_LDS
  // Prefetch tile 0 into buffer 0.
#pragma unroll
  for (int i = 0; i < 4; ++i) {
    const int s = tid + i * 256;
    const int row = s >> 4, c4 = (s & 15) * 4;
    async_b128(kbase_ptr + (size_t)row * EMBED + c4, &kt[0][row * 68 + c4]);
  }
#endif

  float sc[16];
  for (int t8 = 0; t8 < 8; ++t8) {
    const int cur = t8 & 1;
#if HAVE_ASYNC_LDS
    WAIT_ASYNC();                 // my async loads for kt[cur] complete
    __syncthreads();              // everyone's loads done + prev compute done
    if (t8 < 7) {
#pragma unroll
      for (int i = 0; i < 4; ++i) {
        const int s = tid + i * 256;
        const int row = s >> 4, c4 = (s & 15) * 4;
        async_b128(kbase_ptr + (size_t)((t8 + 1) * 64 + row) * EMBED + c4,
                   &kt[1 - cur][row * 68 + c4]);
      }
    }
#else
    __syncthreads();              // prev compute done before overwrite
#pragma unroll
    for (int i = 0; i < 4; ++i) {
      const int s = tid + i * 256;
      const int row = s >> 4, c4 = (s & 15) * 4;
      *(float4*)&kt[cur][row * 68 + c4] =
          *(const float4*)(kbase_ptr + (size_t)(t8 * 64 + row) * EMBED + c4);
    }
    __syncthreads();
#endif
    const float* kr0 = &kt[cur][lane * 68];
    const float* kr1 = &kt[cur][(32 + lane) * 68];
    float s00 = 0.f, s01 = 0.f, s02 = 0.f, s03 = 0.f;
    float s10 = 0.f, s11 = 0.f, s12 = 0.f, s13 = 0.f;
#pragma unroll
    for (int d = 0; d < HEAD_DIM; d += 4) {
      const float4 qv01 = *(const float4*)&vq[wave][d];      // Q[d],v[d],Q[d+1],v[d+1]
      const float4 qv23 = *(const float4*)&vq[wave][d + 2];
      const float4 ka = *(const float4*)(kr0 + d);
      const float4 kb = *(const float4*)(kr1 + d);
      s00 += qv01.y * htanh(qv01.x + ka.x);
      s01 += qv01.w * htanh(qv01.z + ka.y);
      s02 += qv23.y * htanh(qv23.x + ka.z);
      s03 += qv23.w * htanh(qv23.z + ka.w);
      s10 += qv01.y * htanh(qv01.x + kb.x);
      s11 += qv01.w * htanh(qv01.z + kb.y);
      s12 += qv23.y * htanh(qv23.x + kb.z);
      s13 += qv23.w * htanh(qv23.z + kb.w);
    }
    sc[t8 * 2 + 0] = (s00 + s01 + s02 + s03) * 0.125f;       // / sqrt(64)
    sc[t8 * 2 + 1] = (s10 + s11 + s12 + s13) * 0.125f;
  }

  // softmax over 512 values held 16-per-lane (wave32 shuffle reductions)
  float m = sc[0];
#pragma unroll
  for (int i = 1; i < 16; ++i) m = fmaxf(m, sc[i]);
  for (int off = 16; off >= 1; off >>= 1) m = fmaxf(m, __shfl_xor(m, off, 32));
  float s = 0.f;
#pragma unroll
  for (int i = 0; i < 16; ++i) { const float e = __expf(sc[i] - m); sc[i] = e; s += e; }
  for (int off = 16; off >= 1; off >>= 1) s += __shfl_xor(s, off, 32);
  const float inv = 1.0f / s;

  float* arow = attn + ((size_t)(bh * SQ + q)) * SK;
#pragma unroll
  for (int i = 0; i < 16; ++i) {
    const int k = (i >> 1) * 64 + (i & 1) * 32 + lane;
    arow[k] = sc[i] * inv;
  }
}

// ---------------------------------------------------------------------------
// attended = attn @ V per (b,h), via full-precision v_wmma_f32_16x16x4_f32.
// Two interleaved accumulators break the 128-deep WMMA dependency chain.
// Result written in [b, q, h*64+d] layout for the output GEMM.
// ---------------------------------------------------------------------------
__global__ __launch_bounds__(128) void attnv_wmma_f32(
    const float* __restrict__ attn, const float* __restrict__ Vb,
    float* __restrict__ Att) {
  const int wave = threadIdx.x >> 5, lane = threadIdx.x & 31;
  const int t  = blockIdx.x * 4 + wave;         // 2048 tiles
  const int bh = t >> 7;
  const int r  = t & 127;
  const int qt = r >> 2, dt = r & 3;
  const int b = bh >> 2, h = bh & 3;
  const int l16 = lane & 15, half = lane >> 4;

  const float* arow = attn + ((size_t)(bh * SQ + qt * 16 + l16)) * SK;
  const float* vcol = Vb + (size_t)(b * SK) * EMBED + h * HEAD_DIM + dt * 16 + l16;

  v8f c0 = {}, c1 = {};
  for (int k0 = 0; k0 < SK; k0 += 8) {
    {
      const int k = k0 + half * 2;
      const float2 pa = *(const float2*)(arow + k);
      v2f a, bb;
      a[0]  = pa.x;
      a[1]  = pa.y;
      bb[0] = vcol[(size_t)k * EMBED];
      bb[1] = vcol[(size_t)(k + 1) * EMBED];
      c0 = __builtin_amdgcn_wmma_f32_16x16x4_f32(false, a, false, bb,
                                                 (short)0, c0, false, false);
    }
    {
      const int k = k0 + 4 + half * 2;
      const float2 pa = *(const float2*)(arow + k);
      v2f a, bb;
      a[0]  = pa.x;
      a[1]  = pa.y;
      bb[0] = vcol[(size_t)k * EMBED];
      bb[1] = vcol[(size_t)(k + 1) * EMBED];
      c1 = __builtin_amdgcn_wmma_f32_16x16x4_f32(false, a, false, bb,
                                                 (short)0, c1, false, false);
    }
  }
#pragma unroll
  for (int rr = 0; rr < 8; ++rr) {
    const int qq = qt * 16 + rr + half * 8;
    Att[((size_t)(b * SQ + qq)) * EMBED + h * HEAD_DIM + dt * 16 + l16] =
        c0[rr] + c1[rr];
  }
}

// ---------------------------------------------------------------------------
extern "C" void kernel_launch(void* const* d_in, const int* in_sizes, int n_in,
                              void* d_out, int out_size, void* d_ws, size_t ws_size,
                              hipStream_t stream) {
  const float* query = (const float*)d_in[0];
  const float* key   = (const float*)d_in[1];
  const float* value = (const float*)d_in[2];
  const float* Wq    = (const float*)d_in[3];
  const float* bq    = (const float*)d_in[4];
  const float* Wk    = (const float*)d_in[5];
  const float* bk    = (const float*)d_in[6];
  const float* Wv    = (const float*)d_in[7];
  const float* bv    = (const float*)d_in[8];
  const float* vpar  = (const float*)d_in[9];
  const float* Wo    = (const float*)d_in[10];
  const float* bo    = (const float*)d_in[11];

  const size_t BUF = (size_t)MROWS * EMBED;     // 524288 floats = 2 MB
  float* Qb  = (float*)d_ws;
  float* Kb  = Qb + BUF;
  float* Vb  = Kb + BUF;
  float* Att = Vb + BUF;

  float* out  = (float*)d_out;                  // [B, SQ, EMBED]
  float* attn = out + BUF;                      // [B, H, SQ, SK]

  // Projections: 2048 16x16 tiles, 4 waves/block.
  gemm_bf16_wmma<<<512, 128, 0, stream>>>(query, Wq, bq, Qb);
  gemm_bf16_wmma<<<512, 128, 0, stream>>>(key,   Wk, bk, Kb);
  gemm_bf16_wmma<<<512, 128, 0, stream>>>(value, Wv, bv, Vb);

  // Additive scores + softmax: (B*H)=16 x (SQ/8)=64 blocks.
  scores_softmax<<<1024, 256, 0, stream>>>(Qb, Kb, vpar, attn);

  // attn @ V: 2048 tiles, 4 waves/block.
  attnv_wmma_f32<<<512, 128, 0, stream>>>(attn, Vb, Att);

  // Output projection into d_out.
  gemm_bf16_wmma<<<512, 128, 0, stream>>>(Att, Wo, bo, out);
}
